// DecoderOnlyTransformer_72129680769247
// MI455X (gfx1250) — matrix operand order repaired and checked by reading, not compiled, not run on hardware
//
#include <hip/hip_runtime.h>
#include <hip/hip_bf16.h>

// ---------------------------------------------------------------------------
// Model dims (fixed by the reference)
// ---------------------------------------------------------------------------
#define TT   4096
#define DD   768
#define HH   12
#define DKH  64
#define DFF_ 3072
#define VOC  32000
#define NL   2

// ---------------------------------------------------------------------------
// vector types
// ---------------------------------------------------------------------------
typedef __attribute__((ext_vector_type(16))) __bf16 v16bf;
typedef __attribute__((ext_vector_type(8)))  float  v8f;
typedef __attribute__((ext_vector_type(4)))  unsigned int v4u;
typedef __attribute__((ext_vector_type(8)))  int    v8i;
typedef __attribute__((ext_vector_type(4)))  int    v4i;

union AFrag { v16bf v; unsigned short u[16]; };
union BFrag { v16bf v; unsigned short u[16]; };

__device__ __forceinline__ unsigned short f2bf(float f) {
    unsigned u = __float_as_uint(f);
    unsigned r = (u + 0x7FFFu + ((u >> 16) & 1u)) >> 16;
    return (unsigned short)r;
}

__device__ __forceinline__ v8f wmma_bf16(v16bf a, v16bf b, v8f c) {
    // D = A(16x32 bf16) * B(32x16 bf16) + C(16x16 f32)
    return __builtin_amdgcn_wmma_f32_16x16x32_bf16(
        false, a, false, b, (short)0, c, false, false);
}

// ---------------------------------------------------------------------------
// Tensor Data Mover: 2D bf16 tile Global -> LDS (ISA ch.8 D# layout).
// ---------------------------------------------------------------------------
#if __has_builtin(__builtin_amdgcn_tensor_load_to_lds)
#define HAVE_TDM 1
__device__ __forceinline__ void tdm_load_2d_bf16(
        unsigned lds_off, const void* gptr,
        unsigned tensor_d0, unsigned tensor_d1,
        unsigned tile_d0, unsigned tile_d1,
        unsigned long long stride0 /* elements */) {
    unsigned long long ga = (unsigned long long)(uintptr_t)gptr;
    v4u g0;
    g0[0] = 1u;                                            // count=1 (valid)
    g0[1] = lds_off;                                       // lds_addr (bytes)
    g0[2] = (unsigned)(ga & 0xFFFFFFFFu);                  // global_addr lo
    g0[3] = (unsigned)((ga >> 32) & 0x01FFFFFFu) | (2u << 30); // hi | type=2
    v8i g1;
    g1[0] = (int)(1u << 16);                               // data_size=1 (2B)
    g1[1] = (int)((tensor_d0 & 0xFFFFu) << 16);            // dim0 lo16
    g1[2] = (int)((tensor_d0 >> 16) | ((tensor_d1 & 0xFFFFu) << 16));
    g1[3] = (int)((tensor_d1 >> 16) | ((tile_d0 & 0xFFFFu) << 16));
    g1[4] = (int)(tile_d1 & 0xFFFFu);                      // tile_dim1, dim2=0
    g1[5] = (int)(stride0 & 0xFFFFFFFFu);                  // dim0_stride lo32
    g1[6] = (int)((stride0 >> 32) & 0xFFFFu);              // hi16, dim1_stride=0
    g1[7] = 0;
    v4i z4 = {0, 0, 0, 0};
    v8i z8 = {0, 0, 0, 0, 0, 0, 0, 0};
    __builtin_amdgcn_tensor_load_to_lds(g0, g1, z4, z4, z8, 0);
}
#else
#define HAVE_TDM 0
#endif

// ---------------------------------------------------------------------------
// fp32 [K,N] -> bf16 TRANSPOSED [N,K]  (weights; 32x32 LDS-tiled)
// ---------------------------------------------------------------------------
__global__ __launch_bounds__(256) void cvt_bf16_t(const float* __restrict__ in,
                                                  unsigned short* __restrict__ out,
                                                  int K, int N) {
    __shared__ float tile[32][33];
    int tx = threadIdx.x & 31, ty = threadIdx.x >> 5;   // 32 x 8
    int nb = blockIdx.x * 32, kb = blockIdx.y * 32;
    for (int i = ty; i < 32; i += 8)
        tile[i][tx] = in[(size_t)(kb + i) * N + nb + tx];
    __syncthreads();
    for (int i = ty; i < 32; i += 8)
        out[(size_t)(nb + i) * K + kb + tx] = f2bf(tile[tx][i]);
}

// ---------------------------------------------------------------------------
// Embedding gather
// ---------------------------------------------------------------------------
__global__ __launch_bounds__(256) void embed_k(const int* __restrict__ tok,
                                               const float* __restrict__ emb,
                                               float* __restrict__ x) {
    int t = blockIdx.x;
    int id = tok[t];
    for (int d = threadIdx.x; d < DD; d += 256)
        x[(size_t)t * DD + d] = emb[(size_t)id * DD + d];
}

// ---------------------------------------------------------------------------
// LayerNorm (fp32 in, bf16 out)
// ---------------------------------------------------------------------------
__global__ __launch_bounds__(256) void layernorm_k(const float* __restrict__ x,
                                                   const float* __restrict__ g,
                                                   const float* __restrict__ b,
                                                   unsigned short* __restrict__ out) {
    int t = blockIdx.x, tid = threadIdx.x;
    __shared__ float red[256];
    const float* xr = x + (size_t)t * DD;

    float s = 0.f;
    for (int d = tid; d < DD; d += 256) s += xr[d];
    red[tid] = s; __syncthreads();
    for (int off = 128; off > 0; off >>= 1) {
        if (tid < off) red[tid] += red[tid + off];
        __syncthreads();
    }
    float mean = red[0] / (float)DD;
    __syncthreads();

    float vs = 0.f;
    for (int d = tid; d < DD; d += 256) { float z = xr[d] - mean; vs += z * z; }
    red[tid] = vs; __syncthreads();
    for (int off = 128; off > 0; off >>= 1) {
        if (tid < off) red[tid] += red[tid + off];
        __syncthreads();
    }
    float rstd = rsqrtf(red[0] / (float)DD + 1e-5f);

    for (int d = tid; d < DD; d += 256)
        out[(size_t)t * DD + d] = f2bf((xr[d] - mean) * rstd * g[d] + b[d]);
}

// ---------------------------------------------------------------------------
// RoPE + QKV split: qkv f32 [T,2304] -> head-major bf16 q/k/v [H][T][64]
// ---------------------------------------------------------------------------
__global__ __launch_bounds__(128) void rope_split_k(const float* __restrict__ qkv,
                                                    unsigned short* __restrict__ qr,
                                                    unsigned short* __restrict__ kr,
                                                    unsigned short* __restrict__ vb) {
    int t = blockIdx.x, tid = threadIdx.x;
    const float* row = qkv + (size_t)t * (3 * DD);
    for (int idx = tid; idx < HH * 32; idx += 128) {
        int h = idx >> 5, i = idx & 31;
        float inv = __powf(10000.0f, -(float)(2 * i) / 64.0f);
        float sn, cs;
        __sincosf((float)t * inv, &sn, &cs);
        const float* qh = row + h * 192;
        float q1 = qh[2 * i], q2 = qh[2 * i + 1];
        float k1 = qh[64 + 2 * i], k2 = qh[64 + 2 * i + 1];
        size_t base = ((size_t)h * TT + t) * DKH;
        qr[base + i]      = f2bf(q1 * cs - q2 * sn);
        qr[base + 32 + i] = f2bf(q1 * sn + q2 * cs);
        kr[base + i]      = f2bf(k1 * cs - k2 * sn);
        kr[base + 32 + i] = f2bf(k1 * sn + k2 * cs);
    }
    for (int idx = tid; idx < HH * DKH; idx += 128) {
        int h = idx >> 6, d = idx & 63;
        vb[((size_t)h * TT + t) * DKH + d] = f2bf(row[h * 192 + 128 + d]);
    }
}

// ---------------------------------------------------------------------------
// Tiled bf16 WMMA GEMM: C[M,N] = A[M,K]*Bt[N,K]^T (+bias)(+relu)(+residual)
// Workgroup tile 128(M) x 128(N), 8 waves as 4Mx2N; each wave owns 32x64
// (2x4 WMMA accumulators) -> 8 WMMAs per 12 ds_load_b128 per K-step.
// Both tiles are contiguous-row TDM descriptors, double-buffered
// (s_wait_tensorcnt 2 == current pair landed while next pair streams).
// Requires M%128==0, N%128==0, K%32==0 (true for all shapes here).
// ---------------------------------------------------------------------------
__global__ __launch_bounds__(256) void gemm_bf16_k(const unsigned short* __restrict__ A,
                                                   const unsigned short* __restrict__ Bt,
                                                   const float* __restrict__ bias,
                                                   const float* residual,
                                                   float* Cf,
                                                   unsigned short* __restrict__ Cb,
                                                   int M, int N, int K, int relu) {
    __shared__ __align__(16) unsigned short As[2][128][32];  // [m][k]
    __shared__ __align__(16) unsigned short Bs[2][128][32];  // [n][k]

    int tid  = threadIdx.x;
    int lane = tid & 31, wave = tid >> 5;
    int wm = wave >> 1, wn = wave & 1;
    int half = lane >> 4, ln = lane & 15;
    size_t row0 = (size_t)blockIdx.y * 128;
    size_t n0   = (size_t)blockIdx.x * 128;

    v8f acc[2][4];
    for (int i = 0; i < 2; ++i)
        for (int j = 0; j < 4; ++j)
            for (int r = 0; r < 8; ++r) acc[i][j][r] = 0.f;

    auto compute_step = [&](const unsigned short (*Am)[32],
                            const unsigned short (*Bm)[32]) {
        AFrag af[2];
        for (int tm = 0; tm < 2; ++tm)
#pragma unroll
            for (int e = 0; e < 16; ++e) {
                int k = (e < 8) ? (8 * half + e) : (8 * half + e + 8);
                af[tm].u[e] = Am[wm * 32 + tm * 16 + ln][k];
            }
        for (int tn = 0; tn < 4; ++tn) {
            BFrag bf_;
#pragma unroll
            for (int e = 0; e < 16; ++e)
                bf_.u[e] = Bm[wn * 64 + tn * 16 + ln][(e < 8) ? (8 * half + e) : (8 * half + e + 8)];
            for (int tm = 0; tm < 2; ++tm)
                acc[tm][tn] = wmma_bf16(af[tm].v, bf_.v, acc[tm][tn]);
        }
    };

    int nk = K / 32;

#if HAVE_TDM
    bool issuer = (__builtin_amdgcn_readfirstlane((unsigned)tid) < 32u);
    unsigned as_lds[2] = {(unsigned)(uintptr_t)&As[0][0][0],
                          (unsigned)(uintptr_t)&As[1][0][0]};
    unsigned bs_lds[2] = {(unsigned)(uintptr_t)&Bs[0][0][0],
                          (unsigned)(uintptr_t)&Bs[1][0][0]};
    auto issue_pair = [&](int k0, int buf) {
        tdm_load_2d_bf16(as_lds[buf], A + row0 * (size_t)K + k0,
                         (unsigned)(K - k0), (unsigned)(M - (int)row0),
                         32u, 128u, (unsigned long long)K);
        tdm_load_2d_bf16(bs_lds[buf], Bt + n0 * (size_t)K + k0,
                         (unsigned)(K - k0), (unsigned)(N - (int)n0),
                         32u, 128u, (unsigned long long)K);
    };
    if (issuer) issue_pair(0, 0);
    for (int ks = 0; ks < nk; ++ks) {
        int cur = ks & 1;
        if (issuer) {
            if (ks + 1 < nk) {
                issue_pair((ks + 1) * 32, cur ^ 1);
                __builtin_amdgcn_s_wait_tensorcnt(2);  // current pair landed
            } else {
                __builtin_amdgcn_s_wait_tensorcnt(0);
            }
        }
        __syncthreads();
        compute_step(As[cur], Bs[cur]);
        __syncthreads();
    }
#else
    for (int ks = 0; ks < nk; ++ks) {
        int k0 = ks * 32;
        for (int j = 0; j < 2; ++j) {       // A tile: 512 uint4, 2/thread
            int li = tid + j * 256;
            int r = li >> 2, c0 = (li & 3) * 8;
            *(uint4*)(&As[0][r][c0]) =
                *(const uint4*)(A + (row0 + r) * (size_t)K + k0 + c0);
        }
        for (int j = 0; j < 2; ++j) {       // B tile: 512 uint4, 2/thread
            int li = tid + j * 256;
            int r = li >> 2, c0 = (li & 3) * 8;
            *(uint4*)(&Bs[0][r][c0]) =
                *(const uint4*)(Bt + (n0 + r) * (size_t)K + k0 + c0);
        }
        __syncthreads();
        compute_step(As[0], Bs[0]);
        __syncthreads();
    }
#endif

    // Epilogue (C layout: M = r + 8*half, N = ln)
    for (int tm = 0; tm < 2; ++tm)
        for (int tn = 0; tn < 4; ++tn)
#pragma unroll
            for (int r = 0; r < 8; ++r) {
                size_t row = row0 + wm * 32 + tm * 16 + half * 8 + r;
                size_t col = n0 + wn * 64 + tn * 16 + ln;
                float v = acc[tm][tn][r];
                if (bias)     v += bias[col];
                if (relu)     v = fmaxf(v, 0.f);
                if (residual) v += residual[row * N + col];
                size_t idx = row * N + col;
                if (Cf) Cf[idx] = v;
                if (Cb) Cb[idx] = f2bf(v);
            }
}

// ---------------------------------------------------------------------------
// Fused causal flash attention, one wave per (qblock16, head).
// K tile via TDM (contiguous rows); V staged TRANSPOSED manually
// (VsT[d][key]) so V-fragments are contiguous b128 reads.
// ---------------------------------------------------------------------------
__global__ __launch_bounds__(32) void attn_fused_k(const unsigned short* __restrict__ qr,
                                                   const unsigned short* __restrict__ kr,
                                                   const unsigned short* __restrict__ vb,
                                                   unsigned short* __restrict__ o) {
    int qb = blockIdx.x, h = blockIdx.y;
    int lane = threadIdx.x;
    int half = lane >> 4, ln = lane & 15;

    __shared__ __align__(16) unsigned short Ks[32][64];   // [key][dk]
    __shared__ __align__(16) unsigned short VsT[64][32];  // [dk][key]
    __shared__ float          Sf[16][32];
    __shared__ unsigned short Pb[16][32];
    __shared__ float          rowscale[16];
    __shared__ float          rowfac[16];

    // Q fragments (A layout: M = ln, contiguous 8+8 in dk)
    AFrag qf[2];
    const unsigned short* qbase = qr + ((size_t)h * TT + qb * 16 + ln) * DKH;
    for (int s = 0; s < 2; ++s)
#pragma unroll
        for (int e = 0; e < 16; ++e) {
            int k = (e < 8) ? (8 * half + e) : (8 * half + e + 8);
            qf[s].u[e] = qbase[s * 32 + k];
        }

    v8f accO[4];
    for (int tn = 0; tn < 4; ++tn)
        for (int r = 0; r < 8; ++r) accO[tn][r] = 0.f;
    float m_i = -3.0e38f, l_i = 0.f;

#if HAVE_TDM
    unsigned ks_lds = (unsigned)(uintptr_t)&Ks[0][0];
#endif

    int qmax = qb * 16 + 15;
    int nkb = qmax / 32 + 1;

    for (int ib = 0; ib < nkb; ++ib) {
        int kb = ib * 32;
#if HAVE_TDM
        // K block 32x64 contiguous rows -> TDM (single-wave workgroup)
        tdm_load_2d_bf16(ks_lds, kr + ((size_t)h * TT + kb) * DKH,
                         64u, (unsigned)(TT - kb), 64u, 32u, 64ull);
        for (int li = lane; li < 256; li += 32) {
            int key = li >> 3, c0 = (li & 7) * 8;
            uint4 vraw =
                *(const uint4*)(vb + ((size_t)h * TT + kb + key) * DKH + c0);
            const unsigned short* u = (const unsigned short*)&vraw;
#pragma unroll
            for (int i = 0; i < 8; ++i)
                VsT[c0 + i][key] = u[i];
        }
        __builtin_amdgcn_s_wait_tensorcnt(0);
#else
        for (int li = lane; li < 256; li += 32) {
            int key = li >> 3, c0 = (li & 7) * 8;
            *(uint4*)(&Ks[key][c0]) =
                *(const uint4*)(kr + ((size_t)h * TT + kb + key) * DKH + c0);
            uint4 vraw =
                *(const uint4*)(vb + ((size_t)h * TT + kb + key) * DKH + c0);
            const unsigned short* u = (const unsigned short*)&vraw;
#pragma unroll
            for (int i = 0; i < 8; ++i)
                VsT[c0 + i][key] = u[i];
        }
#endif
        __syncthreads();

        // Scores: two 16x16 subtiles over 32 keys
        for (int nt = 0; nt < 2; ++nt) {
            v8f s_acc;
            for (int r = 0; r < 8; ++r) s_acc[r] = 0.f;
            for (int s = 0; s < 2; ++s) {
                BFrag bf_;
#pragma unroll
                for (int e = 0; e < 16; ++e)
                    bf_.u[e] = Ks[nt * 16 + ln][s * 32 + ((e < 8) ? (8 * half + e) : (8 * half + e + 8))];
                s_acc = wmma_bf16(qf[s].v, bf_.v, s_acc);
            }
#pragma unroll
            for (int r = 0; r < 8; ++r) {
                int row = half * 8 + r;
                int col = nt * 16 + ln;
                int key = kb + col, qrow = qb * 16 + row;
                float v = s_acc[r] * 0.125f;   // 1/sqrt(64)
                Sf[row][col] = (key <= qrow) ? v : -3.0e38f;
            }
        }
        __syncthreads();

        if (lane < 16) {
            int row = lane;
            float mx = -3.0e38f;
            for (int c = 0; c < 32; ++c) mx = fmaxf(mx, Sf[row][c]);
            float newm = fmaxf(m_i, mx);
            float sc = __expf(m_i - newm);
            float sum = 0.f;
            for (int c = 0; c < 32; ++c) {
                float p = __expf(Sf[row][c] - newm);
                sum += p;
                Pb[row][c] = f2bf(p);
            }
            l_i = l_i * sc + sum;
            m_i = newm;
            rowscale[row] = sc;
        }
        __syncthreads();

        float sc8[8];
        for (int r = 0; r < 8; ++r) sc8[r] = rowscale[half * 8 + r];
        for (int tn = 0; tn < 4; ++tn)
            for (int r = 0; r < 8; ++r) accO[tn][r] *= sc8[r];

        // O += P(16x32) * V(32x64)
        AFrag pf;
#pragma unroll
        for (int e = 0; e < 16; ++e) {
            int k = (e < 8) ? (8 * half + e) : (8 * half + e + 8);
            pf.u[e] = Pb[ln][k];
        }
        for (int tn = 0; tn < 4; ++tn) {
            BFrag bf_;
#pragma unroll
            for (int e = 0; e < 16; ++e)
                bf_.u[e] = VsT[tn * 16 + ln][(e < 8) ? (8 * half + e) : (8 * half + e + 8)];
            accO[tn] = wmma_bf16(pf.v, bf_.v, accO[tn]);
        }
        __syncthreads();
    }

    if (lane < 16) rowfac[lane] = 1.0f / l_i;
    __syncthreads();
    float f8[8];
    for (int r = 0; r < 8; ++r) f8[r] = rowfac[half * 8 + r];
    for (int tn = 0; tn < 4; ++tn)
#pragma unroll
        for (int r = 0; r < 8; ++r) {
            size_t row = (size_t)qb * 16 + half * 8 + r;
            size_t col = (size_t)h * DKH + tn * 16 + ln;
            o[row * DD + col] = f2bf(accO[tn][r] * f8[r]);
        }
}

// ---------------------------------------------------------------------------
// Host orchestration (input order = setup_inputs insertion order)
// ---------------------------------------------------------------------------
extern "C" void kernel_launch(void* const* d_in, const int* in_sizes, int n_in,
                              void* d_out, int out_size, void* d_ws, size_t ws_size,
                              hipStream_t stream) {
    (void)in_sizes; (void)n_in; (void)out_size; (void)ws_size;

    const int* tokens      = (const int*)d_in[0];
    const float* token_emb = (const float*)d_in[1];
    auto blk = [&](int l, int j) -> const float* { return (const float*)d_in[2 + l * 12 + j]; };
    const float* lnf_g  = (const float*)d_in[26];
    const float* lnf_b  = (const float*)d_in[27];
    const float* head_w = (const float*)d_in[28];
    const float* head_b = (const float*)d_in[29];
    float* out = (float*)d_out;

    char* ws = (char*)d_ws;
    size_t off = 0;
    auto carve = [&](size_t bytes) -> void* {
        void* p = (void*)(ws + off);
        off = (off + bytes + 255) & ~(size_t)255;
        return p;
    };
    float*          x     = (float*)carve((size_t)TT * DD * 4);
    unsigned short* xn    = (unsigned short*)carve((size_t)TT * DD * 2);
    float*          qkvb  = (float*)carve((size_t)TT * 3 * DD * 4);
    unsigned short* qr    = (unsigned short*)carve((size_t)TT * DD * 2);
    unsigned short* kr    = (unsigned short*)carve((size_t)TT * DD * 2);
    unsigned short* vbuf  = (unsigned short*)carve((size_t)TT * DD * 2);
    unsigned short* ao    = (unsigned short*)carve((size_t)TT * DD * 2);
    unsigned short* hbuf  = (unsigned short*)carve((size_t)TT * DD * 2);
    unsigned short* ff1o  = (unsigned short*)carve((size_t)TT * DFF_ * 2);
    unsigned short* w_qkv = (unsigned short*)carve((size_t)DD * 3 * DD * 2);   // [3D][D]
    unsigned short* w_out = (unsigned short*)carve((size_t)DD * DD * 2);       // [D][D]
    unsigned short* w_ff1 = (unsigned short*)carve((size_t)DD * DFF_ * 2);     // [DFF][D]
    unsigned short* w_ff2 = (unsigned short*)carve((size_t)DFF_ * DD * 2);     // [D][DFF]
    unsigned short* w_hd  = (unsigned short*)carve((size_t)DD * VOC * 2);      // [VOC][D]

    // fp32 [K,N] -> bf16 transposed [N,K]
    auto cvt_t = [&](const float* src, unsigned short* dst, int K, int N) {
        cvt_bf16_t<<<dim3(N / 32, K / 32), 256, 0, stream>>>(src, dst, K, N);
    };

    embed_k<<<TT, 256, 0, stream>>>(tokens, token_emb, x);

    for (int l = 0; l < NL; ++l) {
        const float *ln1_g = blk(l,0), *ln1_b = blk(l,1);
        const float *qkv_w = blk(l,2), *qkv_bi = blk(l,3);
        const float *out_w = blk(l,4), *out_bi = blk(l,5);
        const float *ln2_g = blk(l,6), *ln2_b = blk(l,7);
        const float *ff1_w = blk(l,8), *ff1_bi = blk(l,9);
        const float *ff2_w = blk(l,10), *ff2_bi = blk(l,11);

        cvt_t(qkv_w, w_qkv, DD, 3 * DD);
        cvt_t(out_w, w_out, DD, DD);
        cvt_t(ff1_w, w_ff1, DD, DFF_);
        cvt_t(ff2_w, w_ff2, DFF_, DD);

        layernorm_k<<<TT, 256, 0, stream>>>(x, ln1_g, ln1_b, xn);
        gemm_bf16_k<<<dim3(3 * DD / 128, TT / 128), 256, 0, stream>>>(
            xn, w_qkv, qkv_bi, nullptr, qkvb, nullptr, TT, 3 * DD, DD, 0);
        rope_split_k<<<TT, 128, 0, stream>>>(qkvb, qr, kr, vbuf);
        attn_fused_k<<<dim3(TT / 16, HH), 32, 0, stream>>>(qr, kr, vbuf, ao);
        gemm_bf16_k<<<dim3(DD / 128, TT / 128), 256, 0, stream>>>(
            ao, w_out, out_bi, x, x, nullptr, TT, DD, DD, 0);

        layernorm_k<<<TT, 256, 0, stream>>>(x, ln2_g, ln2_b, hbuf);
        gemm_bf16_k<<<dim3(DFF_ / 128, TT / 128), 256, 0, stream>>>(
            hbuf, w_ff1, ff1_bi, nullptr, nullptr, ff1o, TT, DFF_, DD, 1);
        gemm_bf16_k<<<dim3(DD / 128, TT / 128), 256, 0, stream>>>(
            ff1o, w_ff2, ff2_bi, x, x, nullptr, TT, DD, DFF_, 0);
    }

    cvt_t(head_w, w_hd, DD, VOC);
    layernorm_k<<<TT, 256, 0, stream>>>(x, lnf_g, lnf_b, xn);
    gemm_bf16_k<<<dim3(VOC / 128, TT / 128), 256, 0, stream>>>(
        xn, w_hd, head_b, nullptr, out, nullptr, TT, VOC, DD, 0);
}
